// FeatureAdaption_17248588661534
// MI455X (gfx1250) — matrix operand order repaired and checked by reading, not compiled, not run on hardware
//
#include <hip/hip_runtime.h>

typedef __attribute__((ext_vector_type(2)))  float        v2f;
typedef __attribute__((ext_vector_type(8)))  float        v8f;
typedef __attribute__((ext_vector_type(4)))  unsigned int v4u;
typedef __attribute__((ext_vector_type(16))) __bf16       v16bf;

namespace {
constexpr int Bn = 4, Cin = 128, Cout = 128, Hn = 128, Wn = 128;
constexpr int Gn = 4, KKn = 9, Pad = 1;
constexpr int TM = 16;            // pixels per tile (GEMM M)
constexpr int KG = 32 * KKn;      // K per group chunk = 288 (multiple of 32)
constexpr int KTOT = Cin * KKn;   // 1152
constexpr int CPITCH = 18;        // LDS pitch for C transpose (bank-safe)
constexpr int WELEMS = Cout * KTOT;  // 147456 packed bf16 elements per plane
}

__device__ __forceinline__ unsigned short f32_to_bf16_rne(float f) {
  unsigned int u = __builtin_bit_cast(unsigned int, f);
  u += 0x7FFFu + ((u >> 16) & 1u);
  return (unsigned short)(u >> 16);
}

__device__ __forceinline__ float corner(const float* __restrict__ xc, int y, int x, float w) {
  bool ok = (y >= 0) & (y < Hn) & (x >= 0) & (x < Wn);
  int yi = y < 0 ? 0 : (y > Hn - 1 ? Hn - 1 : y);
  int xi = x < 0 ? 0 : (x > Wn - 1 ? Wn - 1 : x);
  float v = xc[yi * Wn + xi];
  return ok ? w * v : 0.0f;
}

// ---- pre-pass: split w_def (fp32) into bf16 hi/lo planes, pre-swizzled into
//      the V_WMMA_F32_16X16X32_BF16 B-fragment lane layout:
//      fragment of wave `ntile`, k-step `ks`: each lane reads a contiguous 32B
//      run holding K = (lane>>4)*16 + 0..15 (2 bf16/dword) for cout = ntile*16 + (lane&15).
__global__ __launch_bounds__(256) void pack_w_bf16(
    const float* __restrict__ w_def, unsigned short* __restrict__ whi,
    unsigned short* __restrict__ wlo) {
  int gid = blockIdx.x * 256 + threadIdx.x;
  if (gid >= WELEMS) return;
  int n = gid / KTOT, k = gid - n * KTOT;        // n = cout, k = c*9 + kk
  int ntile = n >> 4, nn = n & 15;
  int ks = k >> 5, koff = k & 31;
  int ln = ((koff >> 4) << 4) | nn;              // fragment lane
  int t = koff & 15;                             // element within lane's run
  int us = ((((ntile * 36 + ks) * 32 + ln) * 8 + (t >> 1)) << 1) | (t & 1);
  float v = w_def[gid];
  unsigned short h = f32_to_bf16_rne(v);
  float hf = __builtin_bit_cast(float, (unsigned int)h << 16);
  whi[us] = h;
  wlo[us] = f32_to_bf16_rne(v - hf);
}

__global__ __launch_bounds__(256) void dcn_fused_wmma_bf16x3(
    const float* __restrict__ x, const float* __restrict__ w_off,
    const float* __restrict__ b_off, const unsigned short* __restrict__ whi,
    const unsigned short* __restrict__ wlo, float* __restrict__ out) {
  __shared__ float          s_x[Cin][TM];            // staged input pixels (8.0 KB)
  __shared__ float          s_bw[Gn * KKn * TM][4];  // bilinear weights    (9.2 KB)
  __shared__ int            s_y0[Gn * KKn * TM];     //                     (2.3 KB)
  __shared__ int            s_x0[Gn * KKn * TM];     //                     (2.3 KB)
  __shared__ unsigned short s_Ahi[TM][KG];           // A chunk hi bf16     (9.2 KB)
  __shared__ unsigned short s_Alo[TM][KG];           // A chunk lo bf16     (9.2 KB)
  __shared__ float          s_C[8][TM][CPITCH];      // C transpose scratch (9.2 KB)

  const int tile = blockIdx.x;                  // B*H*(W/16) = 4096 tiles
  const int w0 = (tile & 7) * TM;
  const int h  = (tile >> 3) & (Hn - 1);
  const int b  = tile >> 10;
  const int tid  = threadIdx.x;
  const int lane = tid & 31;
  const int wv   = tid >> 5;                    // wave id -> N-tile (16 couts)

  const float* xb = x + (size_t)b * Cin * Hn * Wn;

  // ---- stage x[b, :, h, w0..w0+15] (coalesced) ----
  for (int i = tid; i < Cin * TM; i += 256) {
    int c = i >> 4, m = i & 15;
    s_x[c][m] = xb[(size_t)c * Hn * Wn + h * Wn + w0 + m];
  }
  __syncthreads();

  // ---- fused offset 1x1 conv -> bilinear sampling tuples per (g,kk,m) ----
  for (int i = tid; i < Gn * KKn * TM; i += 256) {
    int m = i & 15, gk = i >> 4;                // gk = g*9 + kk
    float ay = b_off[gk * 2 + 0];
    float ax = b_off[gk * 2 + 1];
    const float* wyp = w_off + (size_t)(gk * 2 + 0) * Cin;
    const float* wxp = w_off + (size_t)(gk * 2 + 1) * Cin;
    #pragma unroll 8
    for (int c = 0; c < Cin; ++c) {
      float xv = s_x[c][m];
      ay = fmaf(xv, wyp[c], ay);
      ax = fmaf(xv, wxp[c], ax);
    }
    int kk = gk % KKn;
    float py = (float)(h + kk / 3 - Pad) + ay;
    float px = (float)(w0 + m + kk % 3 - Pad) + ax;
    float fy = floorf(py), fx = floorf(px);
    float wy1 = py - fy, wx1 = px - fx;
    float wy0 = 1.0f - wy1, wx0 = 1.0f - wx1;
    s_bw[i][0] = wy0 * wx0; s_bw[i][1] = wy0 * wx1;
    s_bw[i][2] = wy1 * wx0; s_bw[i][3] = wy1 * wx1;
    s_y0[i] = (int)fy; s_x0[i] = (int)fx;
  }
  __syncthreads();

  // ---- GEMM: M16 x N16/wave x K1152 via bf16 hi/lo 3-product split ----
  union AF { v4u q[2]; v16bf h; };
  union BF { v4u q[2]; v16bf h; };
  v8f acc0 = {}, acc1 = {};
  const int mA = lane & 15;                     // A row / C(N) lane index
  const int ko = (lane >> 4) << 4;              // A frag K sub-offset (0 or 16)
  const v4u* __restrict__ pBhi = (const v4u*)whi;
  const v4u* __restrict__ pBlo = (const v4u*)wlo;

  for (int g = 0; g < Gn; ++g) {                // K chunk = one deformable group
    // bilinear-sample A chunk (32 channels x 9 taps x 16 pixels), bf16 split
    for (int i = tid; i < TM * KG; i += 256) {
      int m = i / KG, k = i - m * KG;           // k = cc*9 + kk (local K)
      int cc = k / KKn, kk = k - cc * KKn;
      int ti = (g * KKn + kk) * TM + m;
      const float* xc = xb + (size_t)(g * 32 + cc) * Hn * Wn;
      int y0 = s_y0[ti], x0 = s_x0[ti];
      float v = corner(xc, y0,     x0,     s_bw[ti][0])
              + corner(xc, y0,     x0 + 1, s_bw[ti][1])
              + corner(xc, y0 + 1, x0,     s_bw[ti][2])
              + corner(xc, y0 + 1, x0 + 1, s_bw[ti][3]);
      unsigned short hv = f32_to_bf16_rne(v);
      float hf = __builtin_bit_cast(float, (unsigned int)hv << 16);
      s_Ahi[m][k] = hv;
      s_Alo[m][k] = f32_to_bf16_rne(v - hf);
    }
    __syncthreads();
    #pragma unroll
    for (int ksg = 0; ksg < KG / 32; ++ksg) {   // 9 k-steps of 32
      const int kb = ksg * 32;
      AF ahi, alo;                              // contiguous-K A layout (K = ko+0..15): two 16B runs
      ahi.q[0] = *(const v4u*)&s_Ahi[mA][kb + ko];
      ahi.q[1] = *(const v4u*)&s_Ahi[mA][kb + ko + 8];
      alo.q[0] = *(const v4u*)&s_Alo[mA][kb + ko];
      alo.q[1] = *(const v4u*)&s_Alo[mA][kb + ko + 8];
      BF bhi, blo;                              // pre-swizzled, coalesced 32B/lane
      size_t bi = ((size_t)(wv * 36 + g * 9 + ksg) * 32 + lane) * 2;
      bhi.q[0] = pBhi[bi]; bhi.q[1] = pBhi[bi + 1];
      blo.q[0] = pBlo[bi]; blo.q[1] = pBlo[bi + 1];
      v8f& acc = (ksg & 1) ? acc1 : acc0;       // break WMMA RAW chain
      acc = __builtin_amdgcn_wmma_f32_16x16x32_bf16(false, ahi.h, false, bhi.h,
                                                    (short)0, acc, false, false);
      acc = __builtin_amdgcn_wmma_f32_16x16x32_bf16(false, ahi.h, false, blo.h,
                                                    (short)0, acc, false, false);
      acc = __builtin_amdgcn_wmma_f32_16x16x32_bf16(false, alo.h, false, bhi.h,
                                                    (short)0, acc, false, false);
    }
    __syncthreads();
  }

  // ---- ReLU + transpose C through LDS -> coalesced stores along w ----
  #pragma unroll
  for (int j = 0; j < 8; ++j)                   // D: VGPR j -> M = j + 8*(lane>>4), N = lane&15
    s_C[wv][(lane >> 4) * 8 + j][mA] = acc0[j] + acc1[j];
  __syncthreads();
  float* ob = out + (((size_t)b * Cout + wv * 16) * Hn + h) * Wn + w0;
  #pragma unroll
  for (int r = 0; r < 8; ++r) {
    int nn = r * 2 + (lane >> 4);               // 32 lanes = 2 cout rows x 16 w
    float v = s_C[wv][mA][nn];
    ob[(size_t)nn * Hn * Wn + mA] = v > 0.0f ? v : 0.0f;
  }
}

extern "C" void kernel_launch(void* const* d_in, const int* in_sizes, int n_in,
                              void* d_out, int out_size, void* d_ws, size_t ws_size,
                              hipStream_t stream) {
  const float* x     = (const float*)d_in[0];
  const float* w_off = (const float*)d_in[1];
  const float* b_off = (const float*)d_in[2];
  const float* w_def = (const float*)d_in[3];
  float* out = (float*)d_out;

  unsigned short* whi = (unsigned short*)d_ws;          // 294,912 B
  unsigned short* wlo = whi + WELEMS;                   // 294,912 B  (total < 0.6 MB)

  pack_w_bf16<<<dim3((WELEMS + 255) / 256), 256, 0, stream>>>(w_def, whi, wlo);
  dim3 grid(Bn * Hn * (Wn / TM));                       // 4096 tiles
  dcn_fused_wmma_bf16x3<<<grid, 256, 0, stream>>>(x, w_off, b_off, whi, wlo, out);
}
